// RQBottleneck_59064390254708
// MI455X (gfx1250) — matrix operand order repaired and checked by reading, not compiled, not run on hardware
//
#include <hip/hip_runtime.h>
#include <hip/hip_bf16.h>
#include <math.h>

// ---------------------------------------------------------------------------
// RQ bottleneck for MI455X (gfx1250, wave32, WMMA + async LDS DMA).
//   x:         [8,32,32,256] f32  -> N = 8192 rows, D = 256
//   codebooks: [4,8192,256]  f32  -> 4 depths, K = 8192 codes
// Per depth: argmin_k ( ||c_k||^2 - 2 r . c_k ). 8192x8192x256 GEMM + argmin
// via v_wmma_f32_16x16x32_f16 (inline-0 C seed, ||c||^2 folded in epilogue,
// -2 folded into the f16 A fragments). Codebook pre-converted to f16 once;
// the hot loop stages 32-code tiles with global_load_async_to_lds_b128
// (ASYNCcnt DMA), software-pipelines B fragments, and splits K 4-ways across
// workgroups for occupancy (256 WGs x 8 waves).
// ---------------------------------------------------------------------------

typedef __attribute__((ext_vector_type(16))) _Float16 v16h;
typedef __attribute__((ext_vector_type(8)))  float    v8f;

#define RQ_D      256
#define RQ_K      8192
#define RQ_N      8192
#define RQ_DEPTH  4
#define KT        32              // codes per K-tile (2 x 16-wide WMMA sub-tiles)
#define KSPLIT    4               // K-slices across workgroups
#define KPB       (RQ_K / KSPLIT) // 2048 codes per workgroup
#define NKT       (KPB / KT)      // 64 K-tiles per workgroup
#define LDSTRIDE  264             // padded halfs per LDS code row (528B: 16B aligned, conflict-free)

// --------------------------------------------------------------------- prep
// One wave per code row: cnorm = ||c||^2 (f32) and f16 copy of the codebook.
__global__ __launch_bounds__(256) void rq_prep_kernel(
    const float* __restrict__ cb, _Float16* __restrict__ cbh,
    float* __restrict__ cnorm)
{
  int gwid = (blockIdx.x * 256 + threadIdx.x) >> 5;   // global wave id == row
  int lane = threadIdx.x & 31;
  if (gwid >= RQ_DEPTH * RQ_K) return;
  const float* row  = cb  + (size_t)gwid * RQ_D;
  _Float16*    orow = cbh + (size_t)gwid * RQ_D;
  float s = 0.0f;
  #pragma unroll
  for (int c = 0; c < RQ_D / 32; ++c) {
    float v = row[c * 32 + lane];
    s += v * v;
    orow[c * 32 + lane] = (_Float16)v;
  }
  #pragma unroll
  for (int m = 16; m >= 1; m >>= 1) s += __shfl_xor(s, m, 32);
  if (lane == 0) cnorm[gwid] = s;
}

// ------------------------------------------------------------------- argmin
// Grid (64, KSPLIT). Block = 256 threads = 8 waves; each wave owns 16 rows
// (block: 128 rows) and scans this block's 2048-code K-slice. 32-code f16
// tiles are double-buffered in LDS via async DMA; B fragments are software-
// pipelined so ds loads overlap the 16 WMMAs per tile.
__global__ __launch_bounds__(256) void rq_argmin_kernel(
    const float*    __restrict__ resid,   // [N, D] f32 (x itself at depth 0)
    const _Float16* __restrict__ cbh,     // [K, D] f16 (this depth)
    const float*    __restrict__ cnorm,   // [K]    f32 (this depth)
    float* __restrict__ pval,             // [KSPLIT, N] partial min dist
    int*   __restrict__ pidx)             // [KSPLIT, N] partial argmin
{
  __shared__ _Float16 tile[2][KT * LDSTRIDE];
  __shared__ float    s_cnorm[KPB];

  const int tid    = threadIdx.x;
  const int wave   = tid >> 5;
  const int lane   = tid & 31;
  const int col16  = lane & 15;    // N-column within 16-wide tile (C/D: N = lane%16)
  const int half   = lane >> 4;    // upper half-wave -> rows M+8 / K+16 halves
  const int rowbase = blockIdx.x * 128 + wave * 16;
  const int kslice  = blockIdx.y;
  const int kofs    = kslice * KPB;

  // Stage this slice's cnorm into LDS (8KB).
  for (int i = tid; i < KPB; i += 256) s_cnorm[i] = cnorm[kofs + i];

  // A fragments: 16 rows x 256 (f16, scaled by -2), resident for whole K loop.
  // A 16x32 f16 layout (contiguous, matches B): element e -> K = 16*half+e.
  const float* arow = resid + (size_t)(rowbase + col16) * RQ_D;
  v16h afrag[8];
  #pragma unroll
  for (int f = 0; f < 8; ++f) {
    const int d0 = 32 * f + 16 * half;
    #pragma unroll
    for (int e = 0; e < 8; ++e) {
      afrag[f][e]     = (_Float16)(-2.0f * arow[d0 + e]);
      afrag[f][e + 8] = (_Float16)(-2.0f * arow[d0 + 8 + e]);
    }
  }

  float minval[8];
  int   minidx[8];
  #pragma unroll
  for (int i = 0; i < 8; ++i) { minval[i] = INFINITY; minidx[i] = 0; }

  // Async DMA staging: tile = 32 codes x 256 halfs = 16KB = 1024 b128 chunks;
  // chunk c: row = c>>5, col = (c&31)*8 halfs. Each thread issues 4 per-lane
  // global_load_async_to_lds_b128 (consecutive lanes -> consecutive 16B).
  auto stage = [&](int kt, int b) {
    #pragma unroll
    for (int j = 0; j < 4; ++j) {
      const int c   = tid + 256 * j;
      const int row = c >> 5;
      const int col = (c & 31) * 8;
      const _Float16* src = cbh + (size_t)(kofs + kt * KT + row) * RQ_D + col;
      unsigned dst = (unsigned)(uintptr_t)(&tile[b][0] + row * LDSTRIDE + col);
      asm volatile("global_load_async_to_lds_b128 %0, %1, off"
                   :: "v"(dst), "v"((unsigned long long)(uintptr_t)src)
                   : "memory");
    }
  };

  stage(0, 0);
  asm volatile("s_wait_asynccnt 0x0" ::: "memory");
  __syncthreads();

  union BF { v16h v; uint4 u[2]; };

  for (int kt = 0; kt < NKT; ++kt) {
    const int cur = kt & 1;
    // Kick off DMA for the next tile into the other buffer (overlaps WMMAs).
    if (kt + 1 < NKT) stage(kt + 1, cur ^ 1);

    // Prefetch per-column code norms (in flight during the WMMA chain).
    const int klocal = kt * KT;
    const float cn0 = s_cnorm[klocal + col16];
    const float cn1 = s_cnorm[klocal + 16 + col16];

    // B 32x16 f16 in this code's contiguous k order (as A): lanes 0-15 hold K=0..15, lanes 16-31 K=16..31,
    // N = lane%16 -> per fragment f: 16 contiguous halfs at D = 32f+16*half.
    const _Float16* bt0 = &tile[cur][0] + col16 * LDSTRIDE + 16 * half;
    const _Float16* bt1 = bt0 + 16 * LDSTRIDE;

    BF bf0[2], bf1[2];
    {
      const uint4* p0 = (const uint4*)bt0;
      const uint4* p1 = (const uint4*)bt1;
      bf0[0].u[0] = p0[0]; bf0[0].u[1] = p0[1];
      bf1[0].u[0] = p1[0]; bf1[0].u[1] = p1[1];
    }
    v8f acc0 = {};   // inline-0 C seed on the first WMMA of each chain
    v8f acc1 = {};
    #pragma unroll
    for (int f = 0; f < 8; ++f) {
      if (f + 1 < 8) {   // software-pipeline next B fragments for both subs
        const uint4* p0 = (const uint4*)(bt0 + 32 * (f + 1));
        const uint4* p1 = (const uint4*)(bt1 + 32 * (f + 1));
        bf0[(f + 1) & 1].u[0] = p0[0]; bf0[(f + 1) & 1].u[1] = p0[1];
        bf1[(f + 1) & 1].u[0] = p1[0]; bf1[(f + 1) & 1].u[1] = p1[1];
      }
      acc0 = __builtin_amdgcn_wmma_f32_16x16x32_f16(
          false, afrag[f], false, bf0[f & 1].v, (short)0, acc0, false, false);
      acc1 = __builtin_amdgcn_wmma_f32_16x16x32_f16(
          false, afrag[f], false, bf1[f & 1].v, (short)0, acc1, false, false);
    }

    // Running argmin; dist = -2 r.c + ||c||^2. Check sub0 before sub1 so ties
    // resolve to the lower code index (argmin-first semantics).
    const int k0 = kofs + klocal + col16;
    const int k1 = k0 + 16;
    #pragma unroll
    for (int i = 0; i < 8; ++i) {
      const float d0 = acc0[i] + cn0;
      if (d0 < minval[i]) { minval[i] = d0; minidx[i] = k0; }
      const float d1 = acc1[i] + cn1;
      if (d1 < minval[i]) { minval[i] = d1; minidx[i] = k1; }
    }

    // Own async issues done, then block-wide visibility of the next tile.
    asm volatile("s_wait_asynccnt 0x0" ::: "memory");
    __syncthreads();
  }

  // Cross-lane argmin over the 16 columns (xor masks 1,2,4,8 stay within each
  // 16-lane half; lanes l and l+16 hold different rows). Ties -> lowest index.
  #pragma unroll
  for (int m = 1; m < 16; m <<= 1) {
    #pragma unroll
    for (int i = 0; i < 8; ++i) {
      float ov = __shfl_xor(minval[i], m, 32);
      int   oi = __shfl_xor(minidx[i], m, 32);
      if (ov < minval[i] || (ov == minval[i] && oi < minidx[i])) {
        minval[i] = ov; minidx[i] = oi;
      }
    }
  }
  if (col16 == 0) {
    #pragma unroll
    for (int i = 0; i < 8; ++i) {
      const int row = rowbase + i + 8 * half;
      pval[(size_t)kslice * RQ_N + row] = minval[i];
      pidx[(size_t)kslice * RQ_N + row] = minidx[i];
    }
  }
}

// ------------------------------------------------------------------- update
// One block per row. Combine the KSPLIT argmin partials (uniform -> scalar
// loads), then resid -= q ; agg += q ; deterministic loss partial.
__global__ __launch_bounds__(256) void rq_update_kernel(
    const float* __restrict__ x,
    const float* __restrict__ cb,       // this depth's fp32 codebook
    const float* __restrict__ pval,
    const int*   __restrict__ pidx,
    float* __restrict__ resid,
    float* __restrict__ agg,
    float* __restrict__ loss_part,      // [DEPTH * N]
    float* __restrict__ codes_out,      // [N, DEPTH] as float
    float* __restrict__ quants_out,     // [N, D] written at last depth
    int depth)
{
  const int n = blockIdx.x;
  const int d = threadIdx.x;

  float bv = INFINITY; int bi = 0;
  #pragma unroll
  for (int s = 0; s < KSPLIT; ++s) {
    const float v = pval[(size_t)s * RQ_N + n];
    const int   i = pidx[(size_t)s * RQ_N + n];
    if (v < bv || (v == bv && i < bi)) { bv = v; bi = i; }
  }
  const int code = bi;
  const size_t off = (size_t)n * RQ_D + d;

  const float q  = cb[(size_t)code * RQ_D + d];
  const float xv = x[off];
  const float pr = (depth == 0) ? xv   : resid[off];
  const float pa = (depth == 0) ? 0.0f : agg[off];
  const float r = pr - q;
  const float a = pa + q;
  resid[off] = r;
  agg[off]   = a;
  if (depth == RQ_DEPTH - 1) quants_out[off] = a;   // STE forward value == agg
  if (d == 0) codes_out[(size_t)n * RQ_DEPTH + depth] = (float)code;

  // block-reduce sum((x - agg)^2)
  float s = (xv - a) * (xv - a);
  #pragma unroll
  for (int m = 16; m >= 1; m >>= 1) s += __shfl_xor(s, m, 32);
  __shared__ float sm[8];
  if ((d & 31) == 0) sm[d >> 5] = s;
  __syncthreads();
  if (d == 0) {
    float t = 0.0f;
    #pragma unroll
    for (int w = 0; w < 8; ++w) t += sm[w];
    loss_part[(size_t)depth * RQ_N + n] = t;
  }
}

// ----------------------------------------------------------------- finalize
__global__ __launch_bounds__(256) void rq_finalize_kernel(
    const float* __restrict__ loss_part, float* __restrict__ out_loss)
{
  float s = 0.0f;
  for (int i = threadIdx.x; i < RQ_DEPTH * RQ_N; i += 256) s += loss_part[i];
  #pragma unroll
  for (int m = 16; m >= 1; m >>= 1) s += __shfl_xor(s, m, 32);
  __shared__ float sm[8];
  if ((threadIdx.x & 31) == 0) sm[threadIdx.x >> 5] = s;
  __syncthreads();
  if (threadIdx.x == 0) {
    float t = 0.0f;
    #pragma unroll
    for (int w = 0; w < 8; ++w) t += sm[w];
    out_loss[0] = t * (1.0f / (4.0f * 2097152.0f));
  }
}

// ----------------------------------------------------------------- launcher
extern "C" void kernel_launch(void* const* d_in, const int* in_sizes, int n_in,
                              void* d_out, int out_size, void* d_ws, size_t ws_size,
                              hipStream_t stream) {
  const float* x   = (const float*)d_in[0];   // [8,32,32,256]
  const float* cbs = (const float*)d_in[1];   // [4,8192,256]

  float* out        = (float*)d_out;
  float* quants_out = out;                          // 2097152
  float* loss_out   = out + (size_t)RQ_N * RQ_D;    // 1
  float* codes_out  = loss_out + 1;                 // 32768

  char* ws = (char*)d_ws;
  float*    resid = (float*)ws;     ws += (size_t)RQ_N * RQ_D * 4;               // 8 MB
  float*    agg   = (float*)ws;     ws += (size_t)RQ_N * RQ_D * 4;               // 8 MB
  _Float16* cbh   = (_Float16*)ws;  ws += (size_t)RQ_DEPTH * RQ_K * RQ_D * 2;    // 16 MB
  float*    cnorm = (float*)ws;     ws += (size_t)RQ_DEPTH * RQ_K * 4;           // 128 KB
  float*    pval  = (float*)ws;     ws += (size_t)KSPLIT * RQ_N * 4;             // 128 KB
  int*      pidx  = (int*)ws;       ws += (size_t)KSPLIT * RQ_N * 4;             // 128 KB
  float*    lpart = (float*)ws;     ws += (size_t)RQ_DEPTH * RQ_N * 4;           // 128 KB

  rq_prep_kernel<<<(RQ_DEPTH * RQ_K) / 8, 256, 0, stream>>>(cbs, cbh, cnorm);

  dim3 agrid(RQ_N / 128, KSPLIT);
  for (int depth = 0; depth < RQ_DEPTH; ++depth) {
    const float*    rin     = (depth == 0) ? x : resid;
    const float*    cb_d    = cbs   + (size_t)depth * RQ_K * RQ_D;
    const _Float16* cbh_d   = cbh   + (size_t)depth * RQ_K * RQ_D;
    const float*    cnorm_d = cnorm + (size_t)depth * RQ_K;
    rq_argmin_kernel<<<agrid, 256, 0, stream>>>(rin, cbh_d, cnorm_d, pval, pidx);
    rq_update_kernel<<<RQ_N, 256, 0, stream>>>(x, cb_d, pval, pidx, resid, agg,
                                               lpart, codes_out, quants_out, depth);
  }
  rq_finalize_kernel<<<1, 256, 0, stream>>>(lpart, loss_out);
}
